// GenActivation_23527830847979
// MI455X (gfx1250) — matrix operand order repaired and checked by reading, not compiled, not run on hardware
//
#include <hip/hip_runtime.h>
#include <hip/hip_bf16.h>
#include <math.h>

// ---------------- problem constants (from reference) ----------------
#define B_ROWS   16384
#define IN_DIM   2048
#define N_COLS   20
#define MODES    50
#define SPAN     51            // 1 tanh + 50 gumbel logits
#define OUT_DIM  1020          // N_COLS * SPAN
#define N_PAD    1024          // padded N (16 supertiles of 64)
#define TAU_INV  5.0f          // 1 / 0.2
#define EPS      1e-9f

typedef __bf16 v16bf __attribute__((ext_vector_type(16)));
typedef __bf16 v8bf  __attribute__((ext_vector_type(8)));
typedef __bf16 v4bf  __attribute__((ext_vector_type(4)));
typedef float  v8f   __attribute__((ext_vector_type(8)));
typedef float  v4f   __attribute__((ext_vector_type(4)));

#define TILE_K     32
#define COL_STRIDE 40   // bf16 per LDS column: 32 K + 8 pad -> 80 B (16B aligned,
                        // 20-dword bank step: conflict-free over 16 lanes)
#define SLAB_ELEMS (64 * TILE_K)   // 2048 bf16 per W slab per plane

// Combine two aligned 8-wide vector loads into one 16-wide fragment.
__device__ __forceinline__ v16bf ld_frag16(const __bf16* p) {
    const v8bf* q = (const v8bf*)p;     // 16-byte aligned by construction
    v8bf a = q[0];
    v8bf b = *(const v8bf*)(p + 16);
    return __builtin_shufflevector(a, b, 0, 1, 2, 3, 4, 5, 6, 7,
                                         8, 9, 10, 11, 12, 13, 14, 15);
}

// =====================================================================
// Pre-pass 1: split x (fp32, [16384][2048]) into bf16 hi/lo planes.
// Saves 16x redundant in-loop conversion (one per N-supertile).
// =====================================================================
__global__ __launch_bounds__(256)
void convert_x_kernel(const float* __restrict__ x,
                      __bf16* __restrict__ Xhi,
                      __bf16* __restrict__ Xlo)
{
    const size_t i4 = ((size_t)blockIdx.x * 256 + threadIdx.x) * 4;
    if (i4 < (size_t)B_ROWS * IN_DIM) {
        const v4f f = *(const v4f*)(x + i4);
        v4bf h, l;
#pragma unroll
        for (int j = 0; j < 4; ++j) {
            const __bf16 hh = (__bf16)f[j];
            h[j] = hh;
            l[j] = (__bf16)(f[j] - (float)hh);
        }
        *(v4bf*)(Xhi + i4) = h;
        *(v4bf*)(Xlo + i4) = l;
    }
}

// =====================================================================
// Pre-pass 2: split W and swizzle into slab-major layout:
//   slab s = (n/64)*64 + (k/32);  within-slab index = (n%64)*32 + (k%32)
// so GEMM staging becomes a straight b128 copy. Padded cols -> 0.
// =====================================================================
__global__ __launch_bounds__(256)
void convert_w_kernel(const float* __restrict__ W,
                      __bf16* __restrict__ Wswhi,
                      __bf16* __restrict__ Wswlo)
{
    const size_t idx = (size_t)blockIdx.x * 256 + threadIdx.x;
    if (idx < (size_t)IN_DIM * N_PAD) {
        const int kglob = (int)(idx >> 10);      // 0..2047
        const int n     = (int)(idx & (N_PAD - 1));
        const float f   = (n < OUT_DIM) ? W[(size_t)kglob * OUT_DIM + n] : 0.0f;
        const __bf16 h  = (__bf16)f;
        const __bf16 l  = (__bf16)(f - (float)h);
        const size_t s  = (size_t)(n >> 6) * 64 + (kglob >> 5);
        const size_t wi = s * SLAB_ELEMS + (size_t)(n & 63) * TILE_K + (kglob & 31);
        Wswhi[wi] = h;
        Wswlo[wi] = l;
    }
}

// =====================================================================
// GEMM: out[r][n] = sum_k x[r][k] * W[k][n] + b[n]
// bf16x2 split precision on the 16-bit matrix pipe:
//   acc += A_hi*B_hi + A_hi*B_lo + A_lo*B_hi   (~16 mantissa bits)
// Block: 256 threads = 8 waves; each wave owns a 32(M) x 64(N) tile
// (8 v8f accumulators, 24 WMMAs per K-step).
// W slab staged in LDS (bf16 hi/lo, bank-staggered), double-buffered.
// PRECONV=true: pure-copy staging from swizzled planes, bf16 A loads.
// =====================================================================
template <bool PRECONV>
__global__ __launch_bounds__(256)
void gemm_bias_bf16x2_kernel(const float* __restrict__ x,
                             const float* __restrict__ W,
                             const __bf16* __restrict__ Xhi,
                             const __bf16* __restrict__ Xlo,
                             const __bf16* __restrict__ Wswhi,
                             const __bf16* __restrict__ Wswlo,
                             const float* __restrict__ bias,
                             float* __restrict__ out)
{
    __shared__ alignas(16) __bf16 BsHi[2][64 * COL_STRIDE];  // 2 x 5 KB
    __shared__ alignas(16) __bf16 BsLo[2][64 * COL_STRIDE];  // 2 x 5 KB

    const int lane = threadIdx.x & 31;
    const int wave = threadIdx.x >> 5;
    const int row0 = (blockIdx.y * 8 + wave) * 32;   // 32-row wave tile
    const int n0   = blockIdx.x * 64;

    // 16-bit A 16x32 layout: lanes 0-15: M=lane, K {0..7}U{16..23};
    //                        lanes 16-31: M=lane-16, K {8..15}U{24..31}
    const int arow  = lane & 15;
    const int ksel  = lane >> 4;
    // B 32x16 (dense): lane<16: N=lane, K {0..7}U{16..23}; lane>=16: N=lane-16, K {8..15}U{24..31}
    const int ccol  = lane & 15;
    const int rbase = ksel * 8;     // C/D: M = rbase + v, N = ccol

    // Stage a 32(K) x 64(N) W slab (bf16 hi/lo) into LDS buffer `buf`.
    auto stage = [&](int kbase, int buf) {
        if constexpr (PRECONV) {
            // straight vector copy: 1 b128 load + 1 b128 store per plane
            const size_t slab = (size_t)blockIdx.x * 64 + (kbase >> 5);
            const int t   = threadIdx.x;
            const int src = t * 8;                            // 8 bf16 = 16 B
            const int dst = (t >> 2) * COL_STRIDE + (t & 3) * 8;
            const v8bf h = *(const v8bf*)(Wswhi + slab * SLAB_ELEMS + src);
            const v8bf l = *(const v8bf*)(Wswlo + slab * SLAB_ELEMS + src);
            *(v8bf*)(&BsHi[buf][dst]) = h;
            *(v8bf*)(&BsLo[buf][dst]) = l;
        } else {
#pragma unroll
            for (int it = 0; it < 8; ++it) {
                const int t    = threadIdx.x + it * 256;   // 2048 elements
                const int k    = t >> 6;
                const int colv = t & 63;
                const int n    = n0 + colv;
                const float f  = (n < OUT_DIM)
                                     ? W[(size_t)(kbase + k) * OUT_DIM + n]
                                     : 0.0f;
                const __bf16 h = (__bf16)f;
                BsHi[buf][colv * COL_STRIDE + k] = h;
                BsLo[buf][colv * COL_STRIDE + k] = (__bf16)(f - (float)h);
            }
        }
    };

    v8f acc[8] = {};   // acc[r*4 + sub]: r in {0,1} (M halves), sub in 0..3

    stage(0, 0);
    __syncthreads();

    int p = 0;
    for (int k0 = 0; k0 < IN_DIM; k0 += TILE_K) {
        if (k0 + TILE_K < IN_DIM) stage(k0 + TILE_K, p ^ 1);

        // ---- A fragments for both 16-row halves ----
        v16bf a_hi[2], a_lo[2];
#pragma unroll
        for (int r = 0; r < 2; ++r) {
            const int grow = row0 + r * 16 + arow;
            if constexpr (PRECONV) {
                const size_t base = (size_t)grow * IN_DIM + k0 + ksel * 8;
                __builtin_prefetch(Xhi + base + 2 * TILE_K, 0, 0);
                // K {ksel*8..+7} and {16+ksel*8..+7}: two 16 B groups
                const v8bf h0 = *(const v8bf*)(Xhi + base);
                const v8bf h1 = *(const v8bf*)(Xhi + base + 16);
                const v8bf l0 = *(const v8bf*)(Xlo + base);
                const v8bf l1 = *(const v8bf*)(Xlo + base + 16);
                a_hi[r] = __builtin_shufflevector(h0, h1, 0, 1, 2, 3, 4, 5, 6, 7,
                                                  8, 9, 10, 11, 12, 13, 14, 15);
                a_lo[r] = __builtin_shufflevector(l0, l1, 0, 1, 2, 3, 4, 5, 6, 7,
                                                  8, 9, 10, 11, 12, 13, 14, 15);
            } else {
                const float* ap = x + (size_t)grow * IN_DIM + k0 + ksel * 8;
#pragma unroll
                for (int i = 0; i < 8; ++i) {
                    const float f0 = ap[i];
                    const float f1 = ap[16 + i];
                    const __bf16 h0 = (__bf16)f0;
                    const __bf16 h1 = (__bf16)f1;
                    a_hi[r][i]     = h0;  a_lo[r][i]     = (__bf16)(f0 - (float)h0);
                    a_hi[r][8 + i] = h1;  a_lo[r][8 + i] = (__bf16)(f1 - (float)h1);
                }
            }
        }

        // ---- 4 N-subtiles x 2 M-halves: 24 WMMAs per K-step ----
        const __bf16* hb = &BsHi[p][0];
        const __bf16* lb = &BsLo[p][0];
#pragma unroll
        for (int sub = 0; sub < 4; ++sub) {
            const int off = (sub * 16 + ccol) * COL_STRIDE + ksel * 8;
            const v16bf b_hi = ld_frag16(hb + off);
            const v16bf b_lo = ld_frag16(lb + off);
#pragma unroll
            for (int r = 0; r < 2; ++r) {
                v8f c = acc[r * 4 + sub];
                c = __builtin_amdgcn_wmma_f32_16x16x32_bf16(false, a_hi[r], false, b_hi,
                                                            (short)0, c, false, false);
                c = __builtin_amdgcn_wmma_f32_16x16x32_bf16(false, a_hi[r], false, b_lo,
                                                            (short)0, c, false, false);
                c = __builtin_amdgcn_wmma_f32_16x16x32_bf16(false, a_lo[r], false, b_hi,
                                                            (short)0, c, false, false);
                acc[r * 4 + sub] = c;
            }
        }

        __syncthreads();   // staging done + all waves finished reading buf p
        p ^= 1;
    }

    // ---- store C + bias (32-bit C/D layout) ----
#pragma unroll
    for (int sub = 0; sub < 4; ++sub) {
        const int n = n0 + sub * 16 + ccol;
        if (n < OUT_DIM) {
            const float bv = bias[n];
#pragma unroll
            for (int r = 0; r < 2; ++r) {
                const v8f c = acc[r * 4 + sub];
#pragma unroll
                for (int v = 0; v < 8; ++v) {
                    out[(size_t)(row0 + r * 16 + rbase + v) * OUT_DIM + n] = c[v] + bv;
                }
            }
        }
    }
}

// =====================================================================
// Epilogue: gumbel noise + per-span softmax (width 50) + tanh alphas.
// One wave per row; lanes cover j and j+32, __shfl_xor reductions
// (wave32-correct). Pure bandwidth.
// =====================================================================
__global__ __launch_bounds__(256)
void gumbel_act_kernel(const float* __restrict__ outp,
                       const float* __restrict__ u,
                       float* __restrict__ dt)
{
    const int lane = threadIdx.x & 31;
    const int wave = threadIdx.x >> 5;
    const int row  = blockIdx.x * 8 + wave;
    if (row >= B_ROWS) return;

    const float* o  = outp + (size_t)row * OUT_DIM;
    const float* ur = u    + (size_t)row * OUT_DIM;
    float*       d  = dt   + (size_t)row * OUT_DIM;

    for (int c = 0; c < N_COLS; ++c) {
        const int st = c * SPAN;
        if (lane == 0) d[st] = tanhf(o[st]);

        const int j0 = lane;
        const int j1 = lane + 32;
        float z0 = -3.0e38f, z1 = -3.0e38f;
        if (j0 < MODES) {
            const float uu = fminf(fmaxf(ur[st + 1 + j0], EPS), 1.0f - EPS);
            const float g  = -__logf(-__logf(uu));
            z0 = (o[st + 1 + j0] + g) * TAU_INV;
        }
        if (j1 < MODES) {
            const float uu = fminf(fmaxf(ur[st + 1 + j1], EPS), 1.0f - EPS);
            const float g  = -__logf(-__logf(uu));
            z1 = (o[st + 1 + j1] + g) * TAU_INV;
        }

        float m = fmaxf(z0, z1);
#pragma unroll
        for (int off = 16; off > 0; off >>= 1)
            m = fmaxf(m, __shfl_xor(m, off, 32));

        const float e0 = (j0 < MODES) ? __expf(z0 - m) : 0.0f;
        const float e1 = (j1 < MODES) ? __expf(z1 - m) : 0.0f;
        float s = e0 + e1;
#pragma unroll
        for (int off = 16; off > 0; off >>= 1)
            s += __shfl_xor(s, off, 32);

        const float inv = __frcp_rn(s);
        if (j0 < MODES) d[st + 1 + j0] = e0 * inv;
        if (j1 < MODES) d[st + 1 + j1] = e1 * inv;
    }
}

// =====================================================================
extern "C" void kernel_launch(void* const* d_in, const int* in_sizes, int n_in,
                              void* d_out, int out_size, void* d_ws, size_t ws_size,
                              hipStream_t stream) {
    const float* x  = (const float*)d_in[0];   // [16384, 2048]
    const float* W  = (const float*)d_in[1];   // [2048, 1020]
    const float* b  = (const float*)d_in[2];   // [1020]
    const float* u  = (const float*)d_in[3];   // [16384, 1020]

    float* outputs = (float*)d_out;                         // [16384, 1020]
    float* data_t  = outputs + (size_t)B_ROWS * OUT_DIM;    // [16384, 1020]

    const size_t xcnt    = (size_t)B_ROWS * IN_DIM;         // 33.5M
    const size_t wswcnt  = (size_t)IN_DIM * N_PAD;          // 2.1M (padded, swizzled)
    const size_t ws_need = (xcnt * 2 + wswcnt * 2) * sizeof(__bf16);  // ~136 MiB
    const bool   preconv = (ws_size >= ws_need);

    dim3 ggrid(16, B_ROWS / (32 * 8));   // (16, 64): N supertiles x M blocks

    if (preconv) {
        __bf16* Xhi   = (__bf16*)d_ws;
        __bf16* Xlo   = Xhi + xcnt;
        __bf16* Wswhi = Xlo + xcnt;
        __bf16* Wswlo = Wswhi + wswcnt;
        convert_x_kernel<<<(int)((xcnt / 4 + 255) / 256), 256, 0, stream>>>(x, Xhi, Xlo);
        convert_w_kernel<<<(int)((wswcnt + 255) / 256), 256, 0, stream>>>(W, Wswhi, Wswlo);
        gemm_bias_bf16x2_kernel<true><<<ggrid, 256, 0, stream>>>(
            x, W, Xhi, Xlo, Wswhi, Wswlo, b, outputs);
    } else {
        gemm_bias_bf16x2_kernel<false><<<ggrid, 256, 0, stream>>>(
            x, W, nullptr, nullptr, nullptr, nullptr, b, outputs);
    }

    gumbel_act_kernel<<<B_ROWS / 8, 256, 0, stream>>>(outputs, u, data_t);
}
